// CAEC_MoE_90331752170103
// MI455X (gfx1250) — compile-verified, matrix operand order checked by reading, not run on hardware
//
#include <hip/hip_runtime.h>

// ---------------------------------------------------------------------------
// CAEC MoE forward for gfx1250 (MI455X).  bf16 WMMA (16x16x32) for all GEMMs,
// LoRA folded into merged per-(expert,band) weights in a WMMA-swizzled layout,
// weight slabs staged into LDS via the Tensor Data Mover (tensor_load_to_lds +
// s_wait_tensorcnt), persistent MLP blocks, exact radix-select top-k,
// atomic scatter combine.  wave32 / WGP model.
// ---------------------------------------------------------------------------

typedef __attribute__((ext_vector_type(8)))  float  v8f;
typedef __attribute__((ext_vector_type(16))) __bf16 v16bf;
typedef __attribute__((ext_vector_type(8)))  __bf16 v8bf;
typedef unsigned int v4u  __attribute__((ext_vector_type(4)));
typedef int          v8i_ __attribute__((ext_vector_type(8)));
typedef int          v4i_ __attribute__((ext_vector_type(4)));

#define C_IN    96
#define C_OUT   96
#define HID     192
#define K1PAD   128
#define N_EXP   8
#define N_BAND  4
#define N_EB    32
#define NCAT    48        // [chW1 (32) | Wg (8) | pad (8)]
#define LSCALE  2.0f      // 16 / 8
#define W1ELEM  (K1PAD * HID)            // 24576
#define W2ELEM  (HID * C_OUT)            // 18432
#define WEFFSZ  (W1ELEM + W2ELEM)        // 43008 bf16 = 86016 B

#if __has_builtin(__builtin_amdgcn_tensor_load_to_lds)
#define HAVE_TDM 1
#else
#define HAVE_TDM 0
#endif

// monotone float->uint key (order preserving)
__device__ __forceinline__ unsigned fkey(float f) {
    unsigned u = __float_as_uint(f);
    return (u & 0x80000000u) ? ~u : (u | 0x80000000u);
}
__device__ __forceinline__ float keyf(unsigned k) {
    return __uint_as_float((k & 0x80000000u) ? (k ^ 0x80000000u) : ~k);
}
__device__ __forceinline__ float gelu_erf(float v) {
    return 0.5f * v * (1.0f + erff(v * 0.70710678118654752f));
}
// swizzled index for B-operand weights: element (k, n) of a KxN matrix
__device__ __forceinline__ size_t sw_idx(int k, int n, int ldn) {
    return ((size_t)(k >> 5) * ldn + n) * 32 + (k & 31);
}

#if HAVE_TDM
// 1-D contiguous global -> LDS DMA through the Tensor Data Mover.
// D# built per CDNA5 ISA 8.3-8.6: group0 {count=1, lds_addr, global_addr,
// type=2}; group1 {data_size=8B, tensor_dim0=tile_dim0=n8, dim0_stride=n8}.
__device__ __forceinline__ void tdm_load_1d(void* lds_dst, const void* glb_src,
                                            unsigned nbytes) {
    unsigned lds_addr = (unsigned)(size_t)lds_dst;     // low 32b = LDS offset
    unsigned long long ga = (unsigned long long)(size_t)glb_src;
    unsigned n8 = nbytes >> 3;                         // 8-byte elements
    v4u g0 = {};
    g0[0] = 1u;                                        // count=1 (valid D#)
    g0[1] = lds_addr;
    g0[2] = (unsigned)(ga & 0xffffffffu);
    g0[3] = (unsigned)((ga >> 32) & 0x01ffffffu) | (2u << 30);  // type=2
    v8i_ g1 = {};
    g1[0] = (int)(3u << 16);                           // data_size = 8 bytes
    g1[1] = (int)((n8 & 0xffffu) << 16);               // tensor_dim0[15:0]
    g1[2] = (int)((n8 >> 16) & 0xffffu);               // tensor_dim0[31:16]
    g1[3] = (int)((n8 & 0xffffu) << 16);               // tile_dim0
    g1[5] = (int)n8;                                   // tensor_dim0_stride lo
    v4i_ gz = {};
#if defined(__clang_major__) && __clang_major__ >= 23
    v8i_ gz8 = {};
    __builtin_amdgcn_tensor_load_to_lds(g0, g1, gz, gz, gz8, 0);
#else
    __builtin_amdgcn_tensor_load_to_lds(g0, g1, gz, gz, 0);
#endif
}
#endif

// ---- WMMA fragment loaders (16x16x32 bf16, wave32 layouts per CDNA5 ISA) ----
// A: 16x32, row m = lane&15; lane<16 holds K {0..7,16..23}, lane>=16 holds
//    K {8..15,24..31}.  From row-major [M][ld] LDS: two 16B ds_load_b128.
__device__ __forceinline__ v16bf load_a_lds(const __bf16* base, int ld,
                                            int m0, int k0, int lane) {
    const __bf16* p = base + (m0 + (lane & 15)) * ld + k0 + ((lane >> 4) << 3);
    v8bf lo = *(const v8bf*)p;
    v8bf hi = *(const v8bf*)(p + 16);
    return __builtin_shufflevector(lo, hi, 0, 1, 2, 3, 4, 5, 6, 7,
                                           8, 9, 10, 11, 12, 13, 14, 15);
}
// B: 32x16 from swizzled weights (global or LDS): two 16B b128 loads per lane.
__device__ __forceinline__ v16bf load_b_sw(const __bf16* Wsw, int ldn,
                                           int k0, int n0, int lane) {
    const __bf16* p = Wsw + ((size_t)(k0 >> 5) * ldn + n0 + (lane & 15)) * 32
                          + ((lane >> 4) << 3);
    v8bf lo = *(const v8bf*)p;
    v8bf hi = *(const v8bf*)(p + 16);
    return __builtin_shufflevector(lo, hi, 0, 1, 2, 3, 4, 5, 6, 7,
                                           8, 9, 10, 11, 12, 13, 14, 15);
}

// ---------------------------------------------------------------------------
// Kernel 0: init radix-select "remaining" counters
// ---------------------------------------------------------------------------
__global__ void k_init(int* kneed, int cap) {
    if (threadIdx.x < N_EXP) kneed[threadIdx.x] = cap;
}

// ---------------------------------------------------------------------------
// Kernel 1: merge LoRA into bf16 weights (swizzled) + stage aux matrices
//   Weff[e,b] = [W1eff 128x192 | W2eff 192x96]   skipW: 96x96
//   Wcat: 96x48 = [chW1 | Wg | 0]  (affinity head fused GEMM)
// ---------------------------------------------------------------------------
__global__ __launch_bounds__(256)
void k_merge(const float* __restrict__ eW1, const float* __restrict__ eA1,
             const float* __restrict__ eB1, const float* __restrict__ eW2,
             const float* __restrict__ eA2, const float* __restrict__ eB2,
             const float* __restrict__ skipW, const float* __restrict__ chW1,
             const float* __restrict__ Wg,
             __bf16* __restrict__ Weff, __bf16* __restrict__ skipWsw,
             __bf16* __restrict__ catsw) {
    int eb = blockIdx.x, e = eb >> 2, b = eb & 3, tid = threadIdx.x;
    __bf16* W1sw = Weff + (size_t)eb * WEFFSZ;
    __bf16* W2sw = W1sw + W1ELEM;
    for (int i = tid; i < K1PAD * HID; i += 256) {
        int c = i / HID, h = i % HID;
        float w = 0.f;
        if (c < C_IN + 1) {
            w = eW1[((size_t)e * (C_IN + 1) + c) * HID + h];
#pragma unroll
            for (int r = 0; r < 8; ++r)
                w += LSCALE * eA1[((size_t)(e * N_BAND + b) * (C_IN + 1) + c) * 8 + r]
                            * eB1[((size_t)(e * N_BAND + b) * 8 + r) * HID + h];
        }
        W1sw[sw_idx(c, h, HID)] = (__bf16)w;
    }
    for (int i = tid; i < HID * C_OUT; i += 256) {
        int h = i / C_OUT, o = i % C_OUT;
        float w = eW2[((size_t)e * HID + h) * C_OUT + o];
#pragma unroll
        for (int r = 0; r < 8; ++r)
            w += LSCALE * eA2[((size_t)(e * N_BAND + b) * HID + h) * 8 + r]
                        * eB2[((size_t)(e * N_BAND + b) * 8 + r) * C_OUT + o];
        W2sw[sw_idx(h, o, C_OUT)] = (__bf16)w;
    }
    if (eb == 0)
        for (int i = tid; i < C_IN * C_OUT; i += 256) {
            int c = i / C_OUT, n = i % C_OUT;
            skipWsw[sw_idx(c, n, C_OUT)] = (__bf16)skipW[i];
        }
    if (eb == 1)
        for (int i = tid; i < C_IN * NCAT; i += 256) {
            int c = i / NCAT, n = i % NCAT;
            float w = 0.f;
            if (n < 32)      w = chW1[c * 32 + n];
            else if (n < 40) w = Wg[c * N_EXP + (n - 32)];
            catsw[sw_idx(c, n, NCAT)] = (__bf16)w;
        }
}

// ---------------------------------------------------------------------------
// Kernel 2: fused affinity head (WMMA):  [H | logits] = x @ [chW1 | Wg]
// then per-token cs = gelu(H+b1)@chW2 + b2, affinity = logits + lam*cs.
// ---------------------------------------------------------------------------
__global__ __launch_bounds__(128)
void k_affinity(const float* __restrict__ x, const __bf16* __restrict__ catsw,
                const float* __restrict__ lam, const float* __restrict__ chb1,
                const float* __restrict__ chW2, const float* __restrict__ chb2,
                float* __restrict__ aff, unsigned* __restrict__ vmaxkey,
                float* __restrict__ csstats, int N) {
    __shared__ alignas(16) __bf16 lx[64 * C_IN];
    __shared__ float Hs[64 * NCAT];
    __shared__ float s_sum, s_sq;
    __shared__ unsigned smax[N_EXP];
    int tid = threadIdx.x;
    size_t t0 = (size_t)blockIdx.x * 64;
    int nrow = (int)(((size_t)N - t0) < 64 ? (size_t)N - t0 : 64);
    if (tid == 0) { s_sum = 0.f; s_sq = 0.f; }
    if (tid < N_EXP) smax[tid] = 0u;
    for (int i = tid; i < 64 * C_IN; i += 128) lx[i] = (__bf16)0.f;
    __syncthreads();
    for (int i = tid; i < nrow * C_IN; i += 128) lx[i] = (__bf16)x[t0 * C_IN + i];
    __syncthreads();

    int wv = tid >> 5, lane = tid & 31, m0 = wv * 16;
#pragma unroll
    for (int nt = 0; nt < NCAT / 16; ++nt) {
        v8f acc = {};
#pragma unroll
        for (int kc = 0; kc < C_IN / 32; ++kc) {
            v16bf a = load_a_lds(lx, C_IN, m0, kc * 32, lane);
            v16bf b = load_b_sw(catsw, NCAT, kc * 32, nt * 16, lane);
            acc = __builtin_amdgcn_wmma_f32_16x16x32_bf16(
                false, a, false, b, (short)0, acc, false, false);
        }
        int n = lane & 15, mo = (lane >> 4) * 8;
#pragma unroll
        for (int j = 0; j < 8; ++j)
            Hs[(m0 + mo + j) * NCAT + nt * 16 + n] = acc[j];
    }
    __syncthreads();

    if (tid < nrow) {
        const float* hr = &Hs[tid * NCAT];
        float cs = chb2[0];
#pragma unroll 4
        for (int j = 0; j < 32; ++j)
            cs += gelu_erf(hr[j] + chb1[j]) * chW2[j];
        float l = lam[0];
        size_t t = t0 + tid;
#pragma unroll
        for (int e = 0; e < N_EXP; ++e) {
            float a = hr[32 + e] + l * cs;
            aff[t * N_EXP + e] = a;
            atomicMax(&smax[e], fkey(a));
        }
        atomicAdd(&s_sum, cs);
        atomicAdd(&s_sq, cs * cs);
    }
    __syncthreads();
    if (tid < N_EXP) atomicMax(&vmaxkey[tid], smax[tid]);
    if (tid == 0) { atomicAdd(&csstats[0], s_sum); atomicAdd(&csstats[1], s_sq); }
}

// ---------------------------------------------------------------------------
// Kernel 3/4: radix select (kth largest per expert), 4 passes of 8 bits
// ---------------------------------------------------------------------------
__global__ __launch_bounds__(256)
void k_radix_hist(const float* __restrict__ aff, const unsigned* __restrict__ prefix,
                  unsigned* __restrict__ hist, int pass, int N) {
    __shared__ unsigned lh[N_EXP * 256];
    int tid = threadIdx.x;
    for (int i = tid; i < N_EXP * 256; i += 256) lh[i] = 0u;
    __syncthreads();
    size_t t = (size_t)blockIdx.x * 256 + tid;
    if (t < (size_t)N) {
        int shift = 24 - 8 * pass;
#pragma unroll
        for (int e = 0; e < N_EXP; ++e) {
            unsigned key = fkey(aff[t * N_EXP + e]);
            unsigned hi = pass ? (key >> (32 - 8 * pass)) : 0u;
            unsigned pf = pass ? prefix[e] : 0u;
            if (hi == pf) atomicAdd(&lh[e * 256 + ((key >> shift) & 255u)], 1u);
        }
    }
    __syncthreads();
    for (int i = tid; i < N_EXP * 256; i += 256)
        if (lh[i]) atomicAdd(&hist[i], lh[i]);
}

__global__ __launch_bounds__(256)
void k_radix_resolve(unsigned* __restrict__ hist, unsigned* __restrict__ prefix,
                     int* __restrict__ kneed) {
    __shared__ unsigned h[256];
    int tid = threadIdx.x;
    for (int e = 0; e < N_EXP; ++e) {
        h[tid] = hist[e * 256 + tid];
        __syncthreads();
        if (tid == 0) {
            int need = kneed[e];
            unsigned cum = 0; int d;
            for (d = 255; d > 0; --d) {
                if (cum + h[d] >= (unsigned)need) break;
                cum += h[d];
            }
            prefix[e] = (prefix[e] << 8) | (unsigned)d;
            kneed[e]  = need - (int)cum;      // after pass 3: #ties to take
        }
        hist[e * 256 + tid] = 0u;             // ready for next pass
        __syncthreads();
    }
}

// ---------------------------------------------------------------------------
// Kernel 5: compaction into (expert,band) buckets + softmax denominator
// ---------------------------------------------------------------------------
__global__ __launch_bounds__(256)
void k_compact(const float* __restrict__ aff, const int* __restrict__ band,
               const unsigned* __restrict__ prefix, const int* __restrict__ kneed,
               int* __restrict__ tiecnt, int* __restrict__ cnt,
               int* __restrict__ sel_tok, float* __restrict__ sel_val,
               const unsigned* __restrict__ vmaxkey, float* __restrict__ sumexp,
               int cap, int N) {
    size_t t = (size_t)blockIdx.x * 256 + threadIdx.x;
    if (t >= (size_t)N) return;
    int b = band[t];
#pragma unroll
    for (int e = 0; e < N_EXP; ++e) {
        float v = aff[t * N_EXP + e];
        unsigned key = fkey(v);
        unsigned thr = prefix[e];
        bool sel = key > thr;
        if (!sel && key == thr)
            sel = (atomicAdd(&tiecnt[e], 1) < kneed[e]);
        if (sel) {
            int slot = atomicAdd(&cnt[e * N_BAND + b], 1);
            size_t s = (size_t)(e * N_BAND + b) * cap + slot;
            sel_tok[s] = (int)t;
            sel_val[s] = v;
            atomicAdd(&sumexp[e], expf(v - keyf(vmaxkey[e])));
        }
    }
}

// ---------------------------------------------------------------------------
// Kernel 6: expert MLP, persistent blocks.  TDM stages the 86KB merged weight
// slab (W1|W2, swizzled) into LDS once per block, then each block loops over
// 64-token tiles of its (expert,band) bucket.  4 wave32s, 16-row M-tiles.
// GEMM1 (K=128,N=192) -> gelu -> GEMM2 (K=192,N=96) -> gated atomic scatter.
// ---------------------------------------------------------------------------
__global__ __launch_bounds__(128)
void k_mlp(const float* __restrict__ x, const int* __restrict__ sel_tok,
           const float* __restrict__ sel_val, const int* __restrict__ cnt,
           const unsigned* __restrict__ vmaxkey, const float* __restrict__ sumexp,
           const __bf16* __restrict__ Weff,
           const float* __restrict__ eb1, const float* __restrict__ eb2,
           float* __restrict__ out_sums, float* __restrict__ w_sums, int cap) {
    __shared__ alignas(16) __bf16 ldsW[WEFFSZ];        // 86016 B
    __shared__ alignas(16) __bf16 ldsX[64 * K1PAD];    // 16384 B
    __shared__ alignas(16) __bf16 ldsH[64 * HID];      // 24576 B
    __shared__ int   tokLds[64];
    __shared__ float gateLds[64];

    int eb = blockIdx.x, e = eb >> 2, bnd = eb & 3;
    int n_eb = cnt[eb];
    int ntile = (n_eb + 63) >> 6;
    int tid = threadIdx.x;
    int wv = tid >> 5, lane = tid & 31;
    int m0 = wv * 16;

    const __bf16* Wg = Weff + (size_t)eb * WEFFSZ;
#if HAVE_TDM
    if (wv == 0)
        tdm_load_1d(ldsW, Wg, WEFFSZ * 2u);            // async DMA, TENSORcnt
    __builtin_amdgcn_s_wait_tensorcnt(0);
#else
    for (int i = tid; i < WEFFSZ / 8; i += 128)
        ((v8bf*)ldsW)[i] = ((const v8bf*)Wg)[i];
#endif
    __syncthreads();

    const __bf16* W1 = ldsW;
    const __bf16* W2 = ldsW + W1ELEM;
    const float*  b1 = eb1 + e * HID;
    const float*  b2 = eb2 + e * C_OUT;

    for (int tile = blockIdx.y; tile < ntile; tile += gridDim.y) {
        int base = tile * 64;
        int valid = n_eb - base; if (valid > 64) valid = 64;

        if (tid < 64) {
            if (tid < valid) {
                size_t s = (size_t)eb * cap + base + tid;
                tokLds[tid]  = sel_tok[s];
                gateLds[tid] = expf(sel_val[s] - keyf(vmaxkey[e])) / sumexp[e];
            } else {
                tokLds[tid] = 0; gateLds[tid] = 0.f;
            }
        }
        for (int i = tid; i < 64 * K1PAD; i += 128) ldsX[i] = (__bf16)0.f;
        __syncthreads();
        for (int i = tid; i < valid * C_IN; i += 128) {
            int r = i / C_IN, c = i % C_IN;
            ldsX[r * K1PAD + c] = (__bf16)x[(size_t)tokLds[r] * C_IN + c];
        }
        if (tid < valid) ldsX[tid * K1PAD + C_IN] = (__bf16)(float)bnd;
        __syncthreads();

        // ---- GEMM1: h = gelu(Xin @ W1eff + b1) ----
#pragma unroll
        for (int nt = 0; nt < HID / 16; ++nt) {
            v8f acc = {};
#pragma unroll
            for (int kc = 0; kc < K1PAD / 32; ++kc) {
                v16bf a = load_a_lds(ldsX, K1PAD, m0, kc * 32, lane);
                v16bf b = load_b_sw(W1, HID, kc * 32, nt * 16, lane);
                acc = __builtin_amdgcn_wmma_f32_16x16x32_bf16(
                    false, a, false, b, (short)0, acc, false, false);
            }
            int n = lane & 15, mo = (lane >> 4) * 8;
#pragma unroll
            for (int j = 0; j < 8; ++j) {
                int m = m0 + mo + j;
                float v = acc[j] + b1[nt * 16 + n];
                ldsH[m * HID + nt * 16 + n] = (__bf16)gelu_erf(v);
            }
        }
        __syncthreads();

        // ---- GEMM2: out = (h @ W2eff + b2) * gate -> atomic scatter ----
#pragma unroll
        for (int nt = 0; nt < C_OUT / 16; ++nt) {
            v8f acc = {};
#pragma unroll
            for (int kc = 0; kc < HID / 32; ++kc) {
                v16bf a = load_a_lds(ldsH, HID, m0, kc * 32, lane);
                v16bf b = load_b_sw(W2, C_OUT, kc * 32, nt * 16, lane);
                acc = __builtin_amdgcn_wmma_f32_16x16x32_bf16(
                    false, a, false, b, (short)0, acc, false, false);
            }
            int n = lane & 15, mo = (lane >> 4) * 8;
#pragma unroll
            for (int j = 0; j < 8; ++j) {
                int m = m0 + mo + j;
                if (m < valid) {
                    float v = (acc[j] + b2[nt * 16 + n]) * gateLds[m];
                    atomicAdd(&out_sums[(size_t)tokLds[m] * C_OUT + nt * 16 + n], v);
                }
            }
        }
        if (lane < 16) {
            int m = m0 + lane;
            if (m < valid) atomicAdd(&w_sums[tokLds[m]], gateLds[m]);
        }
        __syncthreads();   // protect ldsX/ldsH/tok/gate before next tile
    }
}

// ---------------------------------------------------------------------------
// Kernel 7: skip GEMM + combine normalization + loss scalar
// ---------------------------------------------------------------------------
__global__ __launch_bounds__(128)
void k_final(const float* __restrict__ x, const __bf16* __restrict__ skipWsw,
             const float* __restrict__ skipb, const float* __restrict__ w_sums,
             const float* __restrict__ csstats, float* __restrict__ out, int N) {
    __shared__ alignas(16) __bf16 lx[64 * C_IN];
    int tid = threadIdx.x;
    size_t t0 = (size_t)blockIdx.x * 64;
    int nrow = (int)((size_t)N - t0 < 64 ? (size_t)N - t0 : 64);
    for (int i = tid; i < 64 * C_IN; i += 128) lx[i] = (__bf16)0.f;
    __syncthreads();
    for (int i = tid; i < nrow * C_IN; i += 128) lx[i] = (__bf16)x[t0 * C_IN + i];
    __syncthreads();

    int wv = tid >> 5, lane = tid & 31, m0 = wv * 16;
#pragma unroll
    for (int nt = 0; nt < C_OUT / 16; ++nt) {
        v8f acc = {};
#pragma unroll
        for (int kc = 0; kc < C_IN / 32; ++kc) {
            v16bf a = load_a_lds(lx, C_IN, m0, kc * 32, lane);
            v16bf b = load_b_sw(skipWsw, C_OUT, kc * 32, nt * 16, lane);
            acc = __builtin_amdgcn_wmma_f32_16x16x32_bf16(
                false, a, false, b, (short)0, acc, false, false);
        }
        int n = lane & 15, mo = (lane >> 4) * 8;
#pragma unroll
        for (int j = 0; j < 8; ++j) {
            int m = m0 + mo + j;
            if (m < nrow) {
                size_t t = t0 + m;
                int c = nt * 16 + n;
                float agg = out[t * C_OUT + c] / (w_sums[t] + 1e-6f);
                out[t * C_OUT + c] = acc[j] + skipb[c] + agg;
            }
        }
    }
    if (blockIdx.x == 0 && tid == 0) {
        float s = csstats[0], q = csstats[1];
        float var = (q - s * s / (float)N) / ((float)N - 1.0f);
        out[(size_t)N * C_OUT] = -logf(var + 1e-8f) * 0.1f;
    }
}

// ---------------------------------------------------------------------------
extern "C" void kernel_launch(void* const* d_in, const int* in_sizes, int n_in,
                              void* d_out, int out_size, void* d_ws, size_t ws_size,
                              hipStream_t stream) {
    const float* x     = (const float*)d_in[0];
    const int*   band  = (const int*)  d_in[1];
    const float* Wg    = (const float*)d_in[2];
    const float* lam   = (const float*)d_in[3];
    const float* chW1  = (const float*)d_in[4];
    const float* chb1  = (const float*)d_in[5];
    const float* chW2  = (const float*)d_in[6];
    const float* chb2  = (const float*)d_in[7];
    const float* skipW = (const float*)d_in[8];
    const float* skipb = (const float*)d_in[9];
    const float* eW1   = (const float*)d_in[10];
    const float* eb1   = (const float*)d_in[11];
    const float* eA1   = (const float*)d_in[12];
    const float* eB1   = (const float*)d_in[13];
    const float* eW2   = (const float*)d_in[14];
    const float* eb2   = (const float*)d_in[15];
    const float* eA2   = (const float*)d_in[16];
    const float* eB2   = (const float*)d_in[17];
    (void)n_in; (void)ws_size;

    const int N   = in_sizes[0] / C_IN;
    const int cap = (int)(((long long)N * 5) / 32);   // int(N/8 * 1.25)

    // ---- carve workspace ----
    char* ws = (char*)d_ws;
    size_t off = 0;
    auto carve = [&](size_t bytes) -> char* {
        char* p = ws + off;
        off = (off + bytes + 255) & ~(size_t)255;
        return p;
    };
    float*    aff     = (float*)   carve((size_t)N * N_EXP * 4);
    char*     ctrl0   = ws + off;
    unsigned* hist    = (unsigned*)carve(N_EXP * 256 * 4);
    unsigned* prefix  = (unsigned*)carve(N_EXP * 4);
    int*      kneed   = (int*)     carve(N_EXP * 4);
    int*      tiecnt  = (int*)     carve(N_EXP * 4);
    unsigned* vmaxkey = (unsigned*)carve(N_EXP * 4);
    float*    sumexp  = (float*)   carve(N_EXP * 4);
    float*    csstats = (float*)   carve(2 * 4);
    int*      cnt     = (int*)     carve(N_EB * 4);
    char*     ctrl1   = ws + off;
    int*      sel_tok = (int*)     carve((size_t)N_EB * cap * 4);
    float*    sel_val = (float*)   carve((size_t)N_EB * cap * 4);
    __bf16*   Weff    = (__bf16*)  carve((size_t)N_EB * WEFFSZ * 2);
    __bf16*   skipWsw = (__bf16*)  carve((size_t)C_IN * C_OUT * 2);
    __bf16*   catsw   = (__bf16*)  carve((size_t)C_IN * NCAT * 2);
    float*    w_sums  = (float*)   carve((size_t)N * 4);

    // ---- per-call re-initialization (graph-capture safe) ----
    hipMemsetAsync(ctrl0, 0, (size_t)(ctrl1 - ctrl0), stream);
    hipMemsetAsync(w_sums, 0, (size_t)N * 4, stream);
    hipMemsetAsync(d_out, 0, (size_t)out_size * sizeof(float), stream);
    k_init<<<1, 32, 0, stream>>>(kneed, cap);

    // ---- pipeline ----
    k_merge<<<N_EB, 256, 0, stream>>>(
        eW1, eA1, eB1, eW2, eA2, eB2, skipW, chW1, Wg,
        Weff, skipWsw, catsw);
    k_affinity<<<(N + 63) / 64, 128, 0, stream>>>(
        x, catsw, lam, chb1, chW2, chb2, aff, vmaxkey, csstats, N);
    for (int p = 0; p < 4; ++p) {
        k_radix_hist<<<(N + 255) / 256, 256, 0, stream>>>(aff, prefix, hist, p, N);
        k_radix_resolve<<<1, 256, 0, stream>>>(hist, prefix, kneed);
    }
    k_compact<<<(N + 255) / 256, 256, 0, stream>>>(
        aff, band, prefix, kneed, tiecnt, cnt, sel_tok, sel_val,
        vmaxkey, sumexp, cap, N);
    dim3 gmlp(N_EB, 64);   // persistent blocks: stride over tiles
    k_mlp<<<gmlp, 128, 0, stream>>>(
        x, sel_tok, sel_val, cnt, vmaxkey, sumexp, Weff,
        eb1, eb2, (float*)d_out, w_sums, cap);
    k_final<<<(N + 63) / 64, 128, 0, stream>>>(
        x, skipWsw, skipb, w_sums, csstats, (float*)d_out, N);
}
